// MultiHeadAtten_62242666053769
// MI455X (gfx1250) — matrix operand order repaired and checked by reading, not compiled, not run on hardware
//
#include <hip/hip_runtime.h>
#include <hip/hip_bf16.h>
#include <cstdint>

#define MODEL_D 1024
#define SEQ_L   2048
#define BATCH   4
#define DHEAD   64
#define BHEADS  64                  // B*H
#define MROWS   (BATCH * SEQ_L)     // 8192
#define LN_EPS  1e-5f

typedef __attribute__((ext_vector_type(16))) __bf16 v16bf;
typedef __attribute__((ext_vector_type(8)))  float  v8f;

union FragAB { uint32_t u[8]; __bf16 h[16]; v16bf bf; };

__device__ __forceinline__ uint16_t f2bf(float a) {
  return __builtin_bit_cast(uint16_t, (__bf16)a);        // native v_cvt path
}

__device__ __forceinline__ v8f wmma_bf16(const FragAB& a, const FragAB& b, v8f c) {
  return __builtin_amdgcn_wmma_f32_16x16x32_bf16(
      /*neg_a=*/false, a.bf, /*neg_b=*/false, b.bf,
      /*c_mod=*/(short)0, c, /*reuse_a=*/false, /*reuse_b=*/false);
}

// ---- fragment loaders (layouts per CDNA5 ISA §7.12.2, wave32) ----
// A 16x32 (16-bit): lane&15 = M, dword i half h -> k = (i&3)*2 + (i>>2)*16 + kh*8 + h
__device__ __forceinline__ void frag_a_f32(FragAB& f, const float* row,
                                           int k0, int kh) {
#pragma unroll
  for (int i = 0; i < 8; ++i) {
    const int k = k0 + ((i & 3) << 1) + ((i >> 2) << 4) + (kh << 3);
    const float2 x = *(const float2*)(row + k);
    f.h[2 * i]     = (__bf16)x.x;
    f.h[2 * i + 1] = (__bf16)x.y;
  }
}
// B 32x16 (16-bit): lane&15 = N, dword i half h -> k = (i&3)*2 + (i>>2)*16 + kh*8 + h
__device__ __forceinline__ void frag_b_f32(FragAB& f, const float* row,
                                           int k0, int kh) {
#pragma unroll
  for (int i = 0; i < 8; ++i) {
    const int k = k0 + ((i & 3) << 1) + ((i >> 2) << 4) + (kh << 3);
    const float2 x = *(const float2*)(row + k);
    f.h[2 * i]     = (__bf16)x.x;
    f.h[2 * i + 1] = (__bf16)x.y;
  }
}
__device__ __forceinline__ void frag_a_bf16(FragAB& f, const uint16_t* row,
                                            int k0, int kh) {
#pragma unroll
  for (int i = 0; i < 8; ++i) {
    const int k = k0 + ((i & 3) << 1) + ((i >> 2) << 4) + (kh << 3);
    f.u[i] = *(const uint32_t*)(row + k);
  }
}
__device__ __forceinline__ void frag_b_bf16(FragAB& f, const uint16_t* row,
                                            int k0, int kh) {
#pragma unroll
  for (int i = 0; i < 8; ++i) {
    const int k = k0 + ((i & 3) << 1) + ((i >> 2) << 4) + (kh << 3);
    f.u[i] = *(const uint32_t*)(row + k);
  }
}

// ---------------------------------------------------------------------------
// C[M,N] = A[M,K] @ W[N,K]^T + bias (f32 in, bf16 out). 32x64 tile per wave.
// vt==1: scatter-store into attention-transposed V layout vT[g][dd][l].
// ---------------------------------------------------------------------------
__global__ void __launch_bounds__(256) k_proj_gemm(
    const float* __restrict__ A, const float* __restrict__ W,
    const float* __restrict__ bias, uint16_t* __restrict__ out, int vt)
{
  const int lane = threadIdx.x & 31;
  const int wid  = blockIdx.x * 8 + (threadIdx.x >> 5);  // 4096 waves
  const int tileM = wid >> 4, tileN = wid & 15;
  const int kh = lane >> 4, lr = lane & 15;
  const int m0 = tileM * 32, n0 = tileN * 64;
  const float* arow0 = A + (size_t)(m0 + lr) * MODEL_D;
  const float* arow1 = arow0 + (size_t)16 * MODEL_D;

  v8f c[2][4] = {};
  for (int k0 = 0; k0 < MODEL_D; k0 += 32) {
    __builtin_prefetch(arow0 + k0 + 64, 0, 1);
    __builtin_prefetch(arow1 + k0 + 64, 0, 1);
    FragAB fa0, fa1;
    frag_a_f32(fa0, arow0, k0, kh);
    frag_a_f32(fa1, arow1, k0, kh);
#pragma unroll
    for (int j = 0; j < 4; ++j) {
      const float* wrow = W + (size_t)(n0 + 16 * j + lr) * MODEL_D;
      FragAB fb;
      frag_b_f32(fb, wrow, k0, kh);
      c[0][j] = wmma_bf16(fa0, fb, c[0][j]);
      c[1][j] = wmma_bf16(fa1, fb, c[1][j]);
    }
  }
#pragma unroll
  for (int j = 0; j < 4; ++j) {
    const int cn = n0 + 16 * j + lr;
    const float bv = bias[cn];
#pragma unroll
    for (int mi = 0; mi < 2; ++mi) {
#pragma unroll
      for (int r = 0; r < 8; ++r) {
        const int m = m0 + 16 * mi + kh * 8 + r;
        const uint16_t val = f2bf(c[mi][j][r] + bv);
        if (vt == 0) {
          out[(size_t)m * MODEL_D + cn] = val;
        } else {
          const uint32_t fidx = (uint32_t)m * MODEL_D + (uint32_t)cn;
          const uint32_t g = fidx >> 17, l = (fidx >> 6) & 2047, dd = fidx & 63;
          out[(size_t)((g << 6) + dd) * SEQ_L + l] = val;
        }
      }
    }
  }
}

// ---------------------------------------------------------------------------
// Fused attention: one WG (8 waves) owns (head g, 16-row q-block).
//   phase 1: S = scale * Q Kt into 128KB LDS f32 tile (wave w: cols w*256..+256)
//   phase 2: row softmax (half-wave per row), normalize in LDS, write P f32 to
//            d_out atten (the single mandatory 1GB HBM stream)
//   phase 3: K-split PV: wave w accumulates ctx partial over K = [w*256,+256),
//            A-fragments ds-read from LDS P (cvt->bf16), B from vT (L2-hot);
//            8 partials reduced through 32KB LDS, ctx stored bf16.
// ---------------------------------------------------------------------------
__global__ void __launch_bounds__(256) k_attention(
    const uint16_t* __restrict__ Q, const uint16_t* __restrict__ Kb,
    const uint16_t* __restrict__ VT, float* __restrict__ Pout,
    uint16_t* __restrict__ C)
{
  __shared__ float s_sc[16 * SEQ_L];     // 128 KB: scores -> exp -> P (f32)
  __shared__ float s_part[8 * 16 * 64];  //  32 KB: per-wave ctx partials

  const int t    = threadIdx.x;
  const int lane = t & 31;
  const int w    = t >> 5;               // wave 0..7
  const int kh   = lane >> 4, lr = lane & 15;
  const int g    = blockIdx.x >> 7;
  const int m0   = (blockIdx.x & 127) * 16;

  const uint16_t* qrow = Q + ((size_t)g * SEQ_L + m0 + lr) * DHEAD;

  // ---- phase 1: scores ----
#pragma unroll
  for (int cc = 0; cc < 4; ++cc) {
    const int n0 = w * 256 + cc * 64;
    v8f c[4] = {};
#pragma unroll
    for (int k0 = 0; k0 < DHEAD; k0 += 32) {
      FragAB fa;
      frag_a_bf16(fa, qrow, k0, kh);
#pragma unroll
      for (int j = 0; j < 4; ++j) {
        const uint16_t* krow = Kb + ((size_t)g * SEQ_L + n0 + 16 * j + lr) * DHEAD;
        FragAB fb;
        frag_b_bf16(fb, krow, k0, kh);
        c[j] = wmma_bf16(fa, fb, c[j]);
      }
    }
#pragma unroll
    for (int j = 0; j < 4; ++j)
#pragma unroll
      for (int r = 0; r < 8; ++r)
        s_sc[(kh * 8 + r) * SEQ_L + n0 + 16 * j + lr] = c[j][r] * 0.125f;
  }
  __syncthreads();

  // ---- phase 2: softmax, half-wave per row (2 rows per wave) ----
  {
    const int row = w * 2 + kh;          // 0..15
    float* srow = s_sc + row * SEQ_L;
    float m = -3.4e38f;
    for (int q = 0; q < 128; ++q) m = fmaxf(m, srow[lr + 16 * q]);
#pragma unroll
    for (int msk = 1; msk < 16; msk <<= 1) m = fmaxf(m, __shfl_xor(m, msk, 32));
    float sum = 0.f;
    for (int q = 0; q < 128; ++q) {
      const int n = lr + 16 * q;
      const float e = __expf(srow[n] - m);
      srow[n] = e;
      sum += e;
    }
#pragma unroll
    for (int msk = 1; msk < 16; msk <<= 1) sum += __shfl_xor(sum, msk, 32);
    const float inv = 1.0f / sum;
    float* prow_out = Pout + ((size_t)g * SEQ_L + m0 + row) * SEQ_L;
    for (int q = 0; q < 128; ++q) {
      const int n = lr + 16 * q;
      const float p = srow[n] * inv;
      srow[n] = p;                       // keep normalized P in LDS for PV
      prow_out[n] = p;                   // the single HBM stream of atten
    }
  }
  __syncthreads();

  // ---- phase 3: PV partial over K slice [w*256, w*256+256) ----
  {
    v8f c[4] = {};
    const float* arow = s_sc + lr * SEQ_L;
    const int kbeg = w * 256, kend = kbeg + 256;
    for (int k0 = kbeg; k0 < kend; k0 += 32) {
      FragAB fa;
      frag_a_f32(fa, arow, k0, kh);      // ds_load + cvt_pk_bf16
#pragma unroll
      for (int j = 0; j < 4; ++j) {
        const uint16_t* vrow = VT + ((size_t)(g * 64 + 16 * j + lr)) * SEQ_L;
        FragAB fb;
        frag_b_bf16(fb, vrow, k0, kh);
        c[j] = wmma_bf16(fa, fb, c[j]);
      }
    }
#pragma unroll
    for (int j = 0; j < 4; ++j)
#pragma unroll
      for (int r = 0; r < 8; ++r)
        s_part[(w * 16 + kh * 8 + r) * 64 + 16 * j + lr] = c[j][r];
  }
  __syncthreads();

  // ---- reduce 8 partials -> ctx bf16 ----
  {
#pragma unroll
    for (int o = t * 4; o < t * 4 + 4; ++o) {
      float sum = 0.f;
#pragma unroll
      for (int wv = 0; wv < 8; ++wv) sum += s_part[wv * 1024 + o];
      const int row = o >> 6, col = o & 63;
      C[((size_t)g * SEQ_L + m0 + row) * DHEAD + col] = f2bf(sum);
    }
  }
}

// ---------------------------------------------------------------------------
// attnout[M,N] = ctx[M,K](bf16) @ Wo[N,K]^T + bo (f32 out). 32x64 per wave.
// ---------------------------------------------------------------------------
__global__ void __launch_bounds__(256) k_oproj(
    const uint16_t* __restrict__ A, const float* __restrict__ W,
    const float* __restrict__ bias, float* __restrict__ out)
{
  const int lane = threadIdx.x & 31;
  const int wid  = blockIdx.x * 8 + (threadIdx.x >> 5);  // 4096 waves
  const int tileM = wid >> 4, tileN = wid & 15;
  const int kh = lane >> 4, lr = lane & 15;
  const int m0 = tileM * 32, n0 = tileN * 64;
  const uint16_t* arow0 = A + (size_t)(m0 + lr) * MODEL_D;
  const uint16_t* arow1 = arow0 + (size_t)16 * MODEL_D;

  v8f c[2][4] = {};
  for (int k0 = 0; k0 < MODEL_D; k0 += 32) {
    __builtin_prefetch(arow0 + k0 + 64, 0, 1);
    FragAB fa0, fa1;
    frag_a_bf16(fa0, arow0, k0, kh);
    frag_a_bf16(fa1, arow1, k0, kh);
#pragma unroll
    for (int j = 0; j < 4; ++j) {
      const float* wrow = W + (size_t)(n0 + 16 * j + lr) * MODEL_D;
      FragAB fb;
      frag_b_f32(fb, wrow, k0, kh);
      c[0][j] = wmma_bf16(fa0, fb, c[0][j]);
      c[1][j] = wmma_bf16(fa1, fb, c[1][j]);
    }
  }
#pragma unroll
  for (int j = 0; j < 4; ++j) {
    const int cn = n0 + 16 * j + lr;
    const float bv = bias[cn];
#pragma unroll
    for (int mi = 0; mi < 2; ++mi) {
#pragma unroll
      for (int r = 0; r < 8; ++r) {
        const int m = m0 + 16 * mi + kh * 8 + r;
        out[(size_t)m * MODEL_D + cn] = c[mi][j][r] + bv;
      }
    }
  }
}

// ---------------------------------------------------------------------------
// x = query + attnout; LayerNorm(x)*gamma + beta. One row (1024) per block.
// ---------------------------------------------------------------------------
__global__ void __launch_bounds__(256) k_residual_ln(
    const float* __restrict__ query, const float* __restrict__ attnout,
    const float* __restrict__ gamma, const float* __restrict__ beta,
    float* __restrict__ out)
{
  __shared__ float rsum[256];
  __shared__ float rsq[256];
  const int row = blockIdx.x;
  const float* q = query   + (size_t)row * MODEL_D;
  const float* a = attnout + (size_t)row * MODEL_D;
  const int t = threadIdx.x;
  float x[4]; float s = 0.f, s2 = 0.f;
#pragma unroll
  for (int i = 0; i < 4; ++i) {
    x[i] = q[t + 256 * i] + a[t + 256 * i];
    s += x[i]; s2 += x[i] * x[i];
  }
  rsum[t] = s; rsq[t] = s2; __syncthreads();
  for (int st = 128; st > 0; st >>= 1) {
    if (t < st) { rsum[t] += rsum[t + st]; rsq[t] += rsq[t + st]; }
    __syncthreads();
  }
  const float mu  = rsum[0] * (1.0f / MODEL_D);
  const float var = rsq[0] * (1.0f / MODEL_D) - mu * mu;
  const float rstd = rsqrtf(var + LN_EPS);
  float* o = out + (size_t)row * MODEL_D;
#pragma unroll
  for (int i = 0; i < 4; ++i) {
    const int cidx = t + 256 * i;
    o[cidx] = (x[i] - mu) * rstd * gamma[cidx] + beta[cidx];
  }
}

extern "C" void kernel_launch(void* const* d_in, const int* in_sizes, int n_in,
                              void* d_out, int out_size, void* d_ws, size_t ws_size,
                              hipStream_t stream) {
  (void)in_sizes; (void)n_in; (void)out_size; (void)ws_size;
  const float* query = (const float*)d_in[0];
  const float* key   = (const float*)d_in[1];
  const float* value = (const float*)d_in[2];
  const float* Wq = (const float*)d_in[3];
  const float* bq = (const float*)d_in[4];
  const float* Wk = (const float*)d_in[5];
  const float* bk = (const float*)d_in[6];
  const float* Wv = (const float*)d_in[7];
  const float* bv = (const float*)d_in[8];
  const float* Wo = (const float*)d_in[9];
  const float* bo = (const float*)d_in[10];
  const float* gamma = (const float*)d_in[11];
  const float* beta  = (const float*)d_in[12];

  // workspace: q,k bf16 + vT bf16 + ctx bf16 (16 MB each) + attnout f32 (32 MB)
  char* ws = (char*)d_ws;
  uint16_t* qb = (uint16_t*)(ws);
  uint16_t* kb = (uint16_t*)(ws + (size_t)16 * 1024 * 1024);
  uint16_t* vT = (uint16_t*)(ws + (size_t)32 * 1024 * 1024);
  uint16_t* cb = (uint16_t*)(ws + (size_t)48 * 1024 * 1024);
  float*    ao = (float*)   (ws + (size_t)64 * 1024 * 1024);

  float* out_f = (float*)d_out;                          // [B,L,D] f32
  float* atten = out_f + (size_t)MROWS * MODEL_D;        // [B*H,L,L] f32

  const dim3 blk(256);
  k_proj_gemm  <<<512,  blk, 0, stream>>>(query, Wq, bq, qb, 0);
  k_proj_gemm  <<<512,  blk, 0, stream>>>(key,   Wk, bk, kb, 0);
  k_proj_gemm  <<<512,  blk, 0, stream>>>(value, Wv, bv, vT, 1);
  k_attention  <<<8192, blk, 0, stream>>>(qb, kb, vT, atten, cb);
  k_oproj      <<<512,  blk, 0, stream>>>(cb, Wo, bo, ao);
  k_residual_ln<<<8192, blk, 0, stream>>>(query, ao, gamma, beta, out_f);
}
